// MultiHeadAttention_32753420599796
// MI455X (gfx1250) — matrix operand order repaired and checked by reading, not compiled, not run on hardware
//
#include <hip/hip_runtime.h>

typedef __bf16 bf16_t;
typedef __attribute__((ext_vector_type(16))) __bf16 v16bf;
typedef __attribute__((ext_vector_type(8)))  __bf16 v8bf;
typedef __attribute__((ext_vector_type(8)))  float  v8f;

#define BATCH 4
#define SLEN  2048
#define EMB   1024
#define NH    16
#define HD    64
#define MROWS (BATCH*SLEN)

// ---------------------------------------------------------------------------
// WMMA helpers (bf16 in, f32 accumulate). D = A(16x32) * B(32x16) + C.
// ---------------------------------------------------------------------------
static __device__ __forceinline__ v8f wmma_bf16(v16bf a, v16bf b, v8f c) {
  return __builtin_amdgcn_wmma_f32_16x16x32_bf16(
      /*neg_a=*/false, a, /*neg_b=*/false, b,
      /*c_mod=*/(short)0, c, /*reuse_a=*/false, /*reuse_b=*/false);
}

static __device__ __forceinline__ v16bf cat8(v8bf lo, v8bf hi) {
  return __builtin_shufflevector(lo, hi, 0,1,2,3,4,5,6,7,8,9,10,11,12,13,14,15);
}

// A fragment 16x32 (MxK), row-major source: base = &A[m0][k0], ld = row stride
// Lane layout (ISA 7.12.2): lanes 0-15 -> M=lane, K={kb..kb+7, kb+16..kb+23},
// kb = 8*(lane>=16).
static __device__ __forceinline__ v16bf load_a_frag(const bf16_t* base, int ld, int lane) {
  int m  = lane & 15;
  int kb = (lane >> 4) * 8;
  const bf16_t* p = base + (size_t)m * ld + kb;
  v8bf lo = *(const v8bf*)p;          // K = kb .. kb+7
  v8bf hi = *(const v8bf*)(p + 16);   // K = kb+16 .. kb+23
  return cat8(lo, hi);
}

// B fragment 32x16 (KxN) where B[k][n] = Wrow[n][k] (column-of-B == contiguous
// row of W). base = &W[n0][k0], ld = row stride of W.
// Lane layout: N = lane&15, K = {kb..kb+7, kb+16..kb+23}, kb = 8*(lane>=16).
static __device__ __forceinline__ v16bf load_b_frag(const bf16_t* base, int ld, int lane) {
  int n  = lane & 15;
  int kb = (lane >> 4) * 8;
  const bf16_t* p = base + (size_t)n * ld + kb;
  v8bf lo = *(const v8bf*)p;
  v8bf hi = *(const v8bf*)(p + 16);
  return cat8(lo, hi);
}

// ---------------------------------------------------------------------------
// CDNA5 async global->LDS copy (ASYNCcnt-tracked, ISA ch.10 / 15.18.3).
// LDS byte offset = low 32 bits of the generic pointer (ISA aperture rule).
// ---------------------------------------------------------------------------
static __device__ __forceinline__ void async_cp16(const bf16_t* g, bf16_t* l) {
  unsigned long long ga = (unsigned long long)g;
  unsigned           lo = (unsigned)(unsigned long long)l;
  asm volatile("global_load_async_to_lds_b128 %0, %1, off"
               :: "v"(lo), "v"(ga) : "memory");
}

static __device__ __forceinline__ void wait_async0() {
  asm volatile("s_wait_asynccnt 0" ::: "memory");
}

// ---------------------------------------------------------------------------
// fp32 -> bf16 convert (grid-stride)
// ---------------------------------------------------------------------------
__global__ void cvt_f32_bf16(const float* __restrict__ in, bf16_t* __restrict__ out,
                             size_t n) {
  size_t i = (size_t)blockIdx.x * blockDim.x + threadIdx.x;
  size_t stride = (size_t)gridDim.x * blockDim.x;
  for (; i < n; i += stride) out[i] = (bf16_t)in[i];
}

// ---------------------------------------------------------------------------
// GEMM: C[m,n] = sum_k A[m,k] * W[n,k] + bias[n]
//   mode 0: bf16 out, row-major [M x N]
//   mode 1: bf16 out, per-head transposed V layout (B,H,D,S)
//   mode 2: f32  out, row-major [M x N]
// Block = 256 threads (8 waves, 2x4). Block tile 128x256, wave tile 64x64.
// Double-buffered LDS staging via async global->LDS copies.
// ---------------------------------------------------------------------------
#define APITCH 40   // 32 + 8 pad: row stride 80B keeps b128 LDS reads conflict-free

__global__ __launch_bounds__(256) void gemm_bf16(const bf16_t* __restrict__ A,
                                                 const bf16_t* __restrict__ W,
                                                 const float* __restrict__ bias,
                                                 void* __restrict__ outp,
                                                 int M, int N, int K, int mode) {
  __shared__ __align__(16) bf16_t sA[2][128 * APITCH];  // 20 KB
  __shared__ __align__(16) bf16_t sB[2][256 * APITCH];  // 40 KB

  const int tid  = threadIdx.x;
  const int lane = tid & 31;
  const int wid  = tid >> 5;
  const int lh   = lane >> 4;
  const int l15  = lane & 15;
  const int m0 = blockIdx.x * 128;
  const int n0 = blockIdx.y * 256;
  const int wmB = (wid & 1) * 64;   // wave tile origin in block tile
  const int wnB = (wid >> 1) * 64;

  const int cr = tid >> 2;          // copy row (0..63)
  const int cc = (tid & 3) * 8;     // copy chunk: 8 elements = 16 B

  v8f acc[4][4] = {};
  const int nk = K / 32;

  // prologue: stage k-step 0 into buffer 0
  {
    const bf16_t* Ag = A + (size_t)m0 * K;
    const bf16_t* Wg = W + (size_t)n0 * K;
#pragma unroll
    for (int i = 0; i < 2; ++i)
      async_cp16(Ag + (size_t)(cr + i * 64) * K + cc, &sA[0][(cr + i * 64) * APITCH + cc]);
#pragma unroll
    for (int i = 0; i < 4; ++i)
      async_cp16(Wg + (size_t)(cr + i * 64) * K + cc, &sB[0][(cr + i * 64) * APITCH + cc]);
  }

  for (int kt = 0; kt < nk; ++kt) {
    const int cur = kt & 1;
    wait_async0();
    __syncthreads();

    if (kt + 1 < nk) {               // stage next k-step into the other buffer
      const bf16_t* Ag = A + (size_t)m0 * K + (kt + 1) * 32;
      const bf16_t* Wg = W + (size_t)n0 * K + (kt + 1) * 32;
#pragma unroll
      for (int i = 0; i < 2; ++i)
        async_cp16(Ag + (size_t)(cr + i * 64) * K + cc,
                   &sA[cur ^ 1][(cr + i * 64) * APITCH + cc]);
#pragma unroll
      for (int i = 0; i < 4; ++i)
        async_cp16(Wg + (size_t)(cr + i * 64) * K + cc,
                   &sB[cur ^ 1][(cr + i * 64) * APITCH + cc]);
    }

    v16bf af[4], bfr[4];
#pragma unroll
    for (int i = 0; i < 4; ++i)
      af[i] = load_a_frag(&sA[cur][(wmB + i * 16) * APITCH], APITCH, lane);
#pragma unroll
    for (int j = 0; j < 4; ++j)
      bfr[j] = load_b_frag(&sB[cur][(wnB + j * 16) * APITCH], APITCH, lane);
#pragma unroll
    for (int i = 0; i < 4; ++i)
#pragma unroll
      for (int j = 0; j < 4; ++j)
        acc[i][j] = wmma_bf16(af[i], bfr[j], acc[i][j]);
  }

#pragma unroll
  for (int i = 0; i < 4; ++i)
#pragma unroll
    for (int j = 0; j < 4; ++j) {
      int n = n0 + wnB + j * 16 + l15;
      float bv = bias[n];
#pragma unroll
      for (int r = 0; r < 8; ++r) {
        int m = m0 + wmB + i * 16 + lh * 8 + r;   // C/D layout: row = 8*half + r
        float v = acc[i][j][r] + bv;
        if (mode == 0) {
          ((bf16_t*)outp)[(size_t)m * N + n] = (bf16_t)v;
        } else if (mode == 1) {
          int bb = m >> 11, ss = m & (SLEN - 1);
          int hh = n >> 6,  dd = n & (HD - 1);
          ((bf16_t*)outp)[((size_t)((bb * NH + hh) * HD + dd)) * SLEN + ss] = (bf16_t)v;
        } else {
          ((float*)outp)[(size_t)m * N + n] = v;
        }
      }
    }
}

// ---------------------------------------------------------------------------
// Flash attention. One wave handles one (b, h, 16-query-row) tile.
// Streams keys in chunks of 32, online softmax, accumulates O (16x64) in f32.
// Block = 128 threads (4 waves).
// ---------------------------------------------------------------------------
__global__ __launch_bounds__(128) void attn_kernel(const bf16_t* __restrict__ Qp,
                                                   const bf16_t* __restrict__ Kp,
                                                   const bf16_t* __restrict__ Vt,
                                                   bf16_t* __restrict__ Ob) {
  __shared__ __align__(16) bf16_t ldsP[4][16][40];   // per-wave 16x32 P tile (+8 pad)

  const int lane = threadIdx.x & 31;
  const int wid  = threadIdx.x >> 5;
  const int lh   = lane >> 4;
  const int l15  = lane & 15;
  const int gw = blockIdx.x * 4 + wid;
  const int qt = gw & 127;             // S/16 = 128 query tiles
  const int h  = (gw >> 7) & (NH - 1);
  const int b  = gw >> 11;
  const float scale = 0.125f;          // 1/sqrt(64)

  // Q fragments: rows qt*16..+15, head cols h*64..+63 (two K=32 chunks)
  const bf16_t* qbase = Qp + (size_t)(b * SLEN + qt * 16) * EMB + h * HD;
  v16bf qf0 = load_a_frag(qbase, EMB, lane);
  v16bf qf1 = load_a_frag(qbase + 32, EMB, lane);

  v8f o[4] = {};                       // O tile 16x64 (4 n-tiles of 16)
  float rmax[8], rsum[8];
#pragma unroll
  for (int r = 0; r < 8; ++r) { rmax[r] = -3.0e38f; rsum[r] = 0.0f; }

  for (int kt = 0; kt < SLEN / 32; ++kt) {
    // ----- scores S = Q K^T (16x32 tile) -----
    v8f s0 = {}, s1 = {};
    const bf16_t* kb0 = Kp + (size_t)(b * SLEN + kt * 32) * EMB + h * HD;
    const bf16_t* kb1 = kb0 + (size_t)16 * EMB;
    s0 = wmma_bf16(qf0, load_b_frag(kb0,      EMB, lane), s0);
    s0 = wmma_bf16(qf1, load_b_frag(kb0 + 32, EMB, lane), s0);
    s1 = wmma_bf16(qf0, load_b_frag(kb1,      EMB, lane), s1);
    s1 = wmma_bf16(qf1, load_b_frag(kb1 + 32, EMB, lane), s1);

    // ----- online softmax (rows live in 16-lane halves; reduce width 16) ---
    float p0[8], p1[8], fac[8];
#pragma unroll
    for (int r = 0; r < 8; ++r) {
      float a = s0[r] * scale, c = s1[r] * scale;
      float v = fmaxf(a, c);
      v = fmaxf(v, __shfl_xor(v, 1, 16));
      v = fmaxf(v, __shfl_xor(v, 2, 16));
      v = fmaxf(v, __shfl_xor(v, 4, 16));
      v = fmaxf(v, __shfl_xor(v, 8, 16));
      float nmax = fmaxf(rmax[r], v);
      fac[r] = __expf(rmax[r] - nmax);
      rmax[r] = nmax;
      p0[r] = __expf(a - nmax);
      p1[r] = __expf(c - nmax);
      float ps = p0[r] + p1[r];
      ps += __shfl_xor(ps, 1, 16);
      ps += __shfl_xor(ps, 2, 16);
      ps += __shfl_xor(ps, 4, 16);
      ps += __shfl_xor(ps, 8, 16);
      rsum[r] = rsum[r] * fac[r] + ps;
    }
#pragma unroll
    for (int dt = 0; dt < 4; ++dt)
#pragma unroll
      for (int r = 0; r < 8; ++r) o[dt][r] *= fac[r];

    // ----- transpose P: C-layout -> A-layout via per-wave LDS tile --------
#pragma unroll
    for (int r = 0; r < 8; ++r) {
      ldsP[wid][lh * 8 + r][l15]      = (bf16_t)p0[r];
      ldsP[wid][lh * 8 + r][16 + l15] = (bf16_t)p1[r];
    }
    v16bf pf = load_a_frag(&ldsP[wid][0][0], 40, lane);  // same-wave DS order

    // ----- O += P V : Vt rows are contiguous along keys --------------------
    const bf16_t* vb = Vt + ((size_t)((b * NH + h) * HD)) * SLEN + kt * 32;
#pragma unroll
    for (int dt = 0; dt < 4; ++dt)
      o[dt] = wmma_bf16(pf, load_b_frag(vb + (size_t)(dt * 16) * SLEN, SLEN, lane), o[dt]);
  }

  // ----- epilogue: normalize and store (B,S,E) with head column offset -----
#pragma unroll
  for (int r = 0; r < 8; ++r) {
    float inv = 1.0f / rsum[r];
    int m = b * SLEN + qt * 16 + lh * 8 + r;
#pragma unroll
    for (int dt = 0; dt < 4; ++dt)
      Ob[(size_t)m * EMB + h * HD + dt * 16 + l15] = (bf16_t)(o[dt][r] * inv);
  }
}

// ---------------------------------------------------------------------------
// Host launcher
// ---------------------------------------------------------------------------
extern "C" void kernel_launch(void* const* d_in, const int* in_sizes, int n_in,
                              void* d_out, int out_size, void* d_ws, size_t ws_size,
                              hipStream_t stream) {
  (void)in_sizes; (void)n_in; (void)out_size; (void)ws_size;

  const float* q   = (const float*)d_in[0];
  const float* k   = (const float*)d_in[1];
  const float* v   = (const float*)d_in[2];
  const float* Wq  = (const float*)d_in[3];
  const float* bq  = (const float*)d_in[4];
  const float* Wk  = (const float*)d_in[5];
  const float* bk  = (const float*)d_in[6];
  const float* Wv  = (const float*)d_in[7];
  const float* bv  = (const float*)d_in[8];
  const float* Wo  = (const float*)d_in[9];
  const float* bo  = (const float*)d_in[10];
  float* out = (float*)d_out;

  const size_t ACT  = (size_t)MROWS * EMB;   // 8 Mi elements
  const size_t WSZ  = (size_t)EMB * EMB;     // 1 Mi elements
  char* ws = (char*)d_ws;
  bf16_t* Xq  = (bf16_t*)ws;             ws += ACT * 2;
  bf16_t* Xk  = (bf16_t*)ws;             ws += ACT * 2;
  bf16_t* Xv  = (bf16_t*)ws;             ws += ACT * 2;
  bf16_t* Wqb = (bf16_t*)ws;             ws += WSZ * 2;
  bf16_t* Wkb = (bf16_t*)ws;             ws += WSZ * 2;
  bf16_t* Wvb = (bf16_t*)ws;             ws += WSZ * 2;
  bf16_t* Wob = (bf16_t*)ws;             ws += WSZ * 2;
  bf16_t* Qp  = (bf16_t*)ws;             ws += ACT * 2;
  bf16_t* Kp  = (bf16_t*)ws;             ws += ACT * 2;
  bf16_t* Vt  = (bf16_t*)ws;             ws += ACT * 2;   // (B,H,D,S)
  bf16_t* Ob  = (bf16_t*)ws;             ws += ACT * 2;

  // 1) fp32 -> bf16
  cvt_f32_bf16<<<2048, 256, 0, stream>>>(q,  Xq,  ACT);
  cvt_f32_bf16<<<2048, 256, 0, stream>>>(k,  Xk,  ACT);
  cvt_f32_bf16<<<2048, 256, 0, stream>>>(v,  Xv,  ACT);
  cvt_f32_bf16<<<512,  256, 0, stream>>>(Wq, Wqb, WSZ);
  cvt_f32_bf16<<<512,  256, 0, stream>>>(Wk, Wkb, WSZ);
  cvt_f32_bf16<<<512,  256, 0, stream>>>(Wv, Wvb, WSZ);
  cvt_f32_bf16<<<512,  256, 0, stream>>>(Wo, Wob, WSZ);

  // 2) projections (x @ W^T + b); block tile 128x256
  dim3 gproj(MROWS / 128, EMB / 256);
  gemm_bf16<<<gproj, 256, 0, stream>>>(Xq, Wqb, bq, Qp, MROWS, EMB, EMB, 0);
  gemm_bf16<<<gproj, 256, 0, stream>>>(Xk, Wkb, bk, Kp, MROWS, EMB, EMB, 0);
  gemm_bf16<<<gproj, 256, 0, stream>>>(Xv, Wvb, bv, Vt, MROWS, EMB, EMB, 1);

  // 3) flash attention: B*H*(S/16) = 8192 waves, 4 waves/block
  attn_kernel<<<(BATCH * NH * (SLEN / 16)) / 4, 128, 0, stream>>>(Qp, Kp, Vt, Ob);

  // 4) output projection, f32 result
  gemm_bf16<<<gproj, 256, 0, stream>>>(Ob, Wob, bo, out, MROWS, EMB, EMB, 2);
}
